// PDEEncoder_17789754540090
// MI455X (gfx1250) — compile-verified, matrix operand order checked by reading, not run on hardware
//
#include <hip/hip_runtime.h>
#include <math.h>

// Graphormer encoder for gfx1250 (MI455X), bf16 WMMA path, pre-converted bf16
// operands; GEMMs use 2m x 4n register tiles (8 WMMA per k-step from 6 frag
// loads) to raise arithmetic intensity vs the L2.
// B=4 N=512 D=768 H=32 HD=24 L=4 F=768
#define B_  4
#define N_  512
#define D_  768
#define H_  32
#define HD_ 24
#define L_  4
#define F_  768
#define M_  (B_ * N_)          // 2048 token rows
#define QSCALE 0.20412414523193154f  // 24^-0.5

typedef __attribute__((ext_vector_type(16))) __bf16 v16bf;
typedef __attribute__((ext_vector_type(8)))  float  v8f;

union Frag16 { v16bf v; unsigned short u[16]; };

// ---- helpers ---------------------------------------------------------------

__device__ __forceinline__ unsigned short f2bf(float f) {
  union { float f; unsigned u; } x; x.f = f;
  unsigned r = x.u + 0x7FFFu + ((x.u >> 16) & 1u);   // round-to-nearest-even
  return (unsigned short)(r >> 16);
}

// load 8 bf16 (16 bytes) fragment elements
__device__ __forceinline__ void ld8h(const unsigned short* p, unsigned short* u) {
  *(uint4*)u = *(const uint4*)p;
}

// A/B 16-bit fragment K-striping (CDNA5 ISA 7.12.2): lane&15 = row/col,
// elements 0..7 <- K = 8*half + 0..7 ; elements 8..15 <- K = 16 + 8*half + 0..7

// Shared GEMM core: block tile 128x64 (4 waves stacked in M), wave tile 32x64.
// Per k-step: 2 A frag-pairs + 4 B frag-pairs (12x b128) feed 8 WMMAs.
__device__ __forceinline__ void gemm_core(const unsigned short* __restrict__ A,
                                          const unsigned short* __restrict__ W,
                                          int arow0, int bcol0, int ln, int half,
                                          v8f acc[2][4]) {
  const unsigned short* ar0 = A + (size_t)(arow0 + ln) * D_;
  const unsigned short* ar1 = A + (size_t)(arow0 + 16 + ln) * D_;
  for (int k0 = 0; k0 < D_; k0 += 32) {
    Frag16 a0, a1;
    ld8h(ar0 + k0 + 8 * half, a0.u);
    ld8h(ar0 + k0 + 16 + 8 * half, a0.u + 8);
    ld8h(ar1 + k0 + 8 * half, a1.u);
    ld8h(ar1 + k0 + 16 + 8 * half, a1.u + 8);
#pragma unroll
    for (int t = 0; t < 4; ++t) {
      Frag16 wf;
      const unsigned short* wp = W + (size_t)(bcol0 + t * 16 + ln) * D_ + k0;
      ld8h(wp + 8 * half, wf.u);
      ld8h(wp + 16 + 8 * half, wf.u + 8);
      acc[0][t] = __builtin_amdgcn_wmma_f32_16x16x32_bf16(
          false, a0.v, false, wf.v, (short)0, acc[0][t], false, false);
      acc[1][t] = __builtin_amdgcn_wmma_f32_16x16x32_bf16(
          false, a1.v, false, wf.v, (short)0, acc[1][t], false, false);
    }
  }
}

__device__ __forceinline__ void zero_acc(v8f acc[2][4]) {
  const v8f z = {0.f, 0.f, 0.f, 0.f, 0.f, 0.f, 0.f, 0.f};
#pragma unroll
  for (int i = 0; i < 2; ++i)
#pragma unroll
    for (int t = 0; t < 4; ++t) acc[i][t] = z;
}

// ---- kernel 0: fp32 -> bf16 elementwise conversion (weights) ---------------

__global__ __launch_bounds__(256) void cvt_bf_kernel(
    const float* __restrict__ s, unsigned short* __restrict__ d, int n4) {
  const int i = blockIdx.x * 256 + threadIdx.x;
  if (i < n4) {
    const float4 v = ((const float4*)s)[i];
    ushort4 o;
    o.x = f2bf(v.x); o.y = f2bf(v.y); o.z = f2bf(v.z); o.w = f2bf(v.w);
    ((ushort4*)d)[i] = o;
  }
}

// ---- kernel 1: node feature embedding (writes fp32 + bf16 shadow) ----------

__global__ __launch_bounds__(256) void embed_kernel(
    const int* __restrict__ node_type, const int* __restrict__ in_deg,
    const int* __restrict__ out_deg,
    const float* __restrict__ node_emb, const float* __restrict__ in_emb,
    const float* __restrict__ out_emb, float* __restrict__ x,
    unsigned short* __restrict__ xbf) {
  const int row = blockIdx.x;                 // [0, B*N)
  const int nt = node_type[row];
  const int id = in_deg[row];
  const int od = out_deg[row];
  for (int d = threadIdx.x; d < D_; d += 256) {
    const float v = node_emb[(size_t)nt * D_ + d] +
                    in_emb[(size_t)id * D_ + d] + out_emb[(size_t)od * D_ + d];
    x[(size_t)row * D_ + d] = v;
    xbf[(size_t)row * D_ + d] = f2bf(v);
  }
}

// ---- kernel 2: GEMM + bias + residual -> fp32 (out-proj, fc2) --------------

__global__ __launch_bounds__(128) void gemm_res_f32(
    const unsigned short* __restrict__ A, const unsigned short* __restrict__ W,
    const float* __restrict__ bias, const float* __restrict__ residual,
    float* __restrict__ out) {
  const int ntiles = D_ >> 6;  // 12
  const int bm = blockIdx.x / ntiles;
  const int bn = blockIdx.x % ntiles;
  const int wave = threadIdx.x >> 5;
  const int lane = threadIdx.x & 31;
  const int half = lane >> 4;
  const int ln = lane & 15;

  v8f acc[2][4];
  zero_acc(acc);
  gemm_core(A, W, bm * 128 + wave * 32, bn * 64, ln, half, acc);

#pragma unroll
  for (int t = 0; t < 4; ++t) {
    const int n = bn * 64 + t * 16 + ln;
    const float bb = bias[n];
#pragma unroll
    for (int mi = 0; mi < 2; ++mi)
#pragma unroll
      for (int r = 0; r < 8; ++r) {
        const int m = bm * 128 + wave * 32 + mi * 16 + half * 8 + r;
        out[(size_t)m * D_ + n] =
            acc[mi][t][r] + bb + residual[(size_t)m * D_ + n];
      }
  }
}

// ---- kernel 3: GEMM + bias + exact GELU -> bf16 (fc1) ----------------------

__global__ __launch_bounds__(128) void gemm_gelu_bf(
    const unsigned short* __restrict__ A, const unsigned short* __restrict__ W,
    const float* __restrict__ bias, unsigned short* __restrict__ out) {
  const int ntiles = D_ >> 6;
  const int bm = blockIdx.x / ntiles;
  const int bn = blockIdx.x % ntiles;
  const int wave = threadIdx.x >> 5;
  const int lane = threadIdx.x & 31;
  const int half = lane >> 4;
  const int ln = lane & 15;

  v8f acc[2][4];
  zero_acc(acc);
  gemm_core(A, W, bm * 128 + wave * 32, bn * 64, ln, half, acc);

#pragma unroll
  for (int t = 0; t < 4; ++t) {
    const int n = bn * 64 + t * 16 + ln;
    const float bb = bias[n];
#pragma unroll
    for (int mi = 0; mi < 2; ++mi)
#pragma unroll
      for (int r = 0; r < 8; ++r) {
        const int m = bm * 128 + wave * 32 + mi * 16 + half * 8 + r;
        float v = acc[mi][t][r] + bb;
        v = 0.5f * v * (1.0f + erff(v * 0.70710678118654752f));
        out[(size_t)m * D_ + n] = f2bf(v);
      }
  }
}

// ---- kernel 4: QKV GEMM, writes bf16 head-padded layouts -------------------
// mode 0: q/k -> dst[b][h][n][hd] (hd padded to 32); mode 1: v -> dst[b][h][hd][n]

__global__ __launch_bounds__(128) void gemm_qkv(
    const unsigned short* __restrict__ A, const unsigned short* __restrict__ W,
    const float* __restrict__ bias, unsigned short* __restrict__ dst,
    int mode, float scale) {
  const int ntiles = D_ >> 6;
  const int bm = blockIdx.x / ntiles;
  const int bn = blockIdx.x % ntiles;
  const int wave = threadIdx.x >> 5;
  const int lane = threadIdx.x & 31;
  const int half = lane >> 4;
  const int ln = lane & 15;

  v8f acc[2][4];
  zero_acc(acc);
  gemm_core(A, W, bm * 128 + wave * 32, bn * 64, ln, half, acc);

#pragma unroll
  for (int t = 0; t < 4; ++t) {
    const int n = bn * 64 + t * 16 + ln;     // output feature in [0,768)
    const int h = n / HD_;
    const int hd = n - h * HD_;
    const float bb = bias[n];
#pragma unroll
    for (int mi = 0; mi < 2; ++mi)
#pragma unroll
      for (int r = 0; r < 8; ++r) {
        const int m = bm * 128 + wave * 32 + mi * 16 + half * 8 + r;
        const int b = m >> 9;                  // / N_
        const int nn = m & (N_ - 1);
        const float v = (acc[mi][t][r] + bb) * scale;
        size_t idx;
        if (mode == 1) idx = ((size_t)(b * H_ + h) * 32 + hd) * N_ + nn;   // vT
        else           idx = ((size_t)(b * H_ + h) * N_ + nn) * 32 + hd;   // q/k
        dst[idx] = f2bf(v);
      }
  }
}

// ---- kernel 5: fused biased attention (one wave = one (b,h,16-row q tile)) -

__global__ __launch_bounds__(32) void attn_kernel(
    const unsigned short* __restrict__ qh, const unsigned short* __restrict__ kh,
    const unsigned short* __restrict__ vT, const int* __restrict__ sp,
    const float* __restrict__ attn_bias, const float* __restrict__ sp_emb,
    const float* __restrict__ sp_emb_rev, unsigned short* __restrict__ attn_out) {
  __shared__ float sS[16 * N_];            // 32 KB scores
  __shared__ unsigned short sP[16 * N_];   // 16 KB bf16 probs

  const int bh = blockIdx.x >> 5;          // 32 q-tiles per (b,h)
  const int qt = blockIdx.x & 31;
  const int b = bh >> 5;                   // H_ = 32
  const int h = bh & 31;
  const int lane = threadIdx.x;
  const int half = lane >> 4;
  const int ln = lane & 15;
  const int qbase = qt * 16;

  // Q fragment: rows qbase..qbase+15, full padded head dim K=32
  Frag16 qa;
  {
    const unsigned short* qr = qh + ((size_t)bh * N_ + qbase + ln) * 32;
    ld8h(qr + 8 * half, qa.u);
    ld8h(qr + 16 + 8 * half, qa.u + 8);
  }

  const v8f z = {0.f, 0.f, 0.f, 0.f, 0.f, 0.f, 0.f, 0.f};

  // S = Q*K^T + bias  -> LDS (fp32)
  for (int kt = 0; kt < 32; ++kt) {
    Frag16 kf;
    const int key = kt * 16 + ln;
    const unsigned short* kr = kh + ((size_t)bh * N_ + key) * 32;
    ld8h(kr + 8 * half, kf.u);
    ld8h(kr + 16 + 8 * half, kf.u + 8);
    v8f s = __builtin_amdgcn_wmma_f32_16x16x32_bf16(
        false, qa.v, false, kf.v, (short)0, z, false, false);
#pragma unroll
    for (int r = 0; r < 8; ++r) {
      const int m = half * 8 + r;
      const int q = qbase + m;
      const int s1 = sp[((size_t)b * N_ + q) * N_ + key];
      const int s2 = sp[((size_t)b * N_ + key) * N_ + q];
      const float bias = sp_emb[s1 * H_ + h] + sp_emb_rev[s2 * H_ + h] +
                         attn_bias[((size_t)b * N_ + q) * N_ + key];
      sS[m * N_ + key] = s[r] + bias;
    }
  }
  __syncthreads();

  // row softmax (fp32) -> bf16 probs in LDS
  for (int m = 0; m < 16; ++m) {
    float mx = -3.0e38f;
    for (int i = lane; i < N_; i += 32) mx = fmaxf(mx, sS[m * N_ + i]);
#pragma unroll
    for (int d = 16; d > 0; d >>= 1) mx = fmaxf(mx, __shfl_xor(mx, d, 32));
    float sum = 0.f;
    for (int i = lane; i < N_; i += 32) {
      const float e = __expf(sS[m * N_ + i] - mx);
      sS[m * N_ + i] = e;
      sum += e;
    }
#pragma unroll
    for (int d = 16; d > 0; d >>= 1) sum += __shfl_xor(sum, d, 32);
    const float inv = 1.0f / sum;
    for (int i = lane; i < N_; i += 32) sP[m * N_ + i] = f2bf(sS[m * N_ + i] * inv);
  }
  __syncthreads();

  // O = P * V   (K over 512 keys, N over padded head dim 32) -> bf16
#pragma unroll
  for (int nt = 0; nt < 2; ++nt) {
    v8f acc = z;
    const int hd = nt * 16 + ln;
    for (int kc = 0; kc < 16; ++kc) {
      Frag16 pa, vf;
      const unsigned short* pr = sP + ln * N_ + kc * 32;
      ld8h(pr + 8 * half, pa.u);
      ld8h(pr + 16 + 8 * half, pa.u + 8);
      const unsigned short* vr = vT + ((size_t)bh * 32 + hd) * N_ + kc * 32;
      ld8h(vr + 8 * half, vf.u);
      ld8h(vr + 16 + 8 * half, vf.u + 8);
      acc = __builtin_amdgcn_wmma_f32_16x16x32_bf16(
          false, pa.v, false, vf.v, (short)0, acc, false, false);
    }
    if (hd < HD_) {
#pragma unroll
      for (int r = 0; r < 8; ++r) {
        const int m = half * 8 + r;
        attn_out[((size_t)b * N_ + qbase + m) * D_ + h * HD_ + hd] = f2bf(acc[r]);
      }
    }
  }
}

// ---- kernel 6: LayerNorm (writes fp32 + bf16 shadow) -----------------------

__global__ __launch_bounds__(256) void ln_kernel(
    const float* __restrict__ t, const float* __restrict__ w,
    const float* __restrict__ bi, float* __restrict__ out,
    unsigned short* __restrict__ outbf) {
  const int row = blockIdx.x;
  __shared__ float s_sum[8], s_sq[8];
  const int wid = threadIdx.x >> 5;
  const int lane = threadIdx.x & 31;
  float sum = 0.f, sq = 0.f;
  for (int i = threadIdx.x; i < D_; i += 256) {
    const float v = t[(size_t)row * D_ + i];
    sum += v; sq += v * v;
  }
#pragma unroll
  for (int d = 16; d > 0; d >>= 1) {
    sum += __shfl_xor(sum, d, 32);
    sq  += __shfl_xor(sq, d, 32);
  }
  if (lane == 0) { s_sum[wid] = sum; s_sq[wid] = sq; }
  __syncthreads();
  float tot = 0.f, totsq = 0.f;
#pragma unroll
  for (int i = 0; i < 8; ++i) { tot += s_sum[i]; totsq += s_sq[i]; }
  const float mu = tot * (1.0f / D_);
  const float var = totsq * (1.0f / D_) - mu * mu;
  const float rstd = rsqrtf(var + 1e-5f);
  for (int i = threadIdx.x; i < D_; i += 256) {
    const float v = (t[(size_t)row * D_ + i] - mu) * rstd * w[i] + bi[i];
    out[(size_t)row * D_ + i] = v;
    outbf[(size_t)row * D_ + i] = f2bf(v);
  }
}

// ---- launcher --------------------------------------------------------------

extern "C" void kernel_launch(void* const* d_in, const int* in_sizes, int n_in,
                              void* d_out, int out_size, void* d_ws, size_t ws_size,
                              hipStream_t stream) {
  const int*   node_type = (const int*)d_in[0];
  const int*   in_degree = (const int*)d_in[1];
  const int*   out_degree= (const int*)d_in[2];
  const int*   spatial   = (const int*)d_in[3];
  const float* attn_bias = (const float*)d_in[4];
  const float* node_emb  = (const float*)d_in[5];
  const float* in_emb    = (const float*)d_in[6];
  const float* out_emb   = (const float*)d_in[7];
  const float* sp_emb    = (const float*)d_in[8];
  const float* sp_emb_rev= (const float*)d_in[9];
  const float* qw = (const float*)d_in[10]; const float* qb = (const float*)d_in[11];
  const float* kw = (const float*)d_in[12]; const float* kb = (const float*)d_in[13];
  const float* vw = (const float*)d_in[14]; const float* vb = (const float*)d_in[15];
  const float* ow = (const float*)d_in[16]; const float* ob = (const float*)d_in[17];
  const float* fc1w = (const float*)d_in[18]; const float* fc1b = (const float*)d_in[19];
  const float* fc2w = (const float*)d_in[20]; const float* fc2b = (const float*)d_in[21];
  const float* ln1w = (const float*)d_in[22]; const float* ln1b = (const float*)d_in[23];
  const float* ln2w = (const float*)d_in[24]; const float* ln2b = (const float*)d_in[25];

  const size_t XB = (size_t)M_ * D_ * sizeof(float);                 // 6 MB
  const size_t XH = (size_t)M_ * D_ * sizeof(unsigned short);        // 3 MB
  const size_t HB = (size_t)B_ * H_ * N_ * 32 * sizeof(unsigned short); // 4 MB
  const size_t WB = (size_t)L_ * D_ * D_ * sizeof(unsigned short);   // 4.7 MB

  char* ws = (char*)d_ws;
  float* x            = (float*)ws;                       ws += XB;
  float* tmp          = (float*)ws;                       ws += XB;
  unsigned short* xbf = (unsigned short*)ws;              ws += XH;
  unsigned short* abf = (unsigned short*)ws;              ws += XH;  // attn out
  unsigned short* hbf = (unsigned short*)ws;              ws += XH;  // ffn hidden
  unsigned short* qh  = (unsigned short*)ws;              ws += HB;
  unsigned short* kh  = (unsigned short*)ws;              ws += HB;
  unsigned short* vT  = (unsigned short*)ws;              ws += HB;
  unsigned short* wqb = (unsigned short*)ws;              ws += WB;
  unsigned short* wkb = (unsigned short*)ws;              ws += WB;
  unsigned short* wvb = (unsigned short*)ws;              ws += WB;
  unsigned short* wob = (unsigned short*)ws;              ws += WB;
  unsigned short* w1b = (unsigned short*)ws;              ws += WB;
  unsigned short* w2b = (unsigned short*)ws;              ws += WB;

  // zero head buffers once per launch so the hd=24..31 pad is always zero
  hipMemsetAsync(qh, 0, 3 * HB, stream);

  // pre-convert all weights to bf16 (elementwise, one float4 per thread, ~84 MB traffic)
  const int n4 = (L_ * D_ * D_) / 4;           // 2,359,296 / 4
  const int gc = (n4 + 255) / 256;
  cvt_bf_kernel<<<gc, 256, 0, stream>>>(qw,   wqb, n4);
  cvt_bf_kernel<<<gc, 256, 0, stream>>>(kw,   wkb, n4);
  cvt_bf_kernel<<<gc, 256, 0, stream>>>(vw,   wvb, n4);
  cvt_bf_kernel<<<gc, 256, 0, stream>>>(ow,   wob, n4);
  cvt_bf_kernel<<<gc, 256, 0, stream>>>(fc1w, w1b, n4);
  cvt_bf_kernel<<<gc, 256, 0, stream>>>(fc2w, w2b, n4);

  embed_kernel<<<M_, 256, 0, stream>>>(node_type, in_degree, out_degree,
                                       node_emb, in_emb, out_emb, x, xbf);

  const dim3 gGemm((M_ / 128) * (D_ / 64));   // 192 blocks
  for (int l = 0; l < L_; ++l) {
    const size_t wo = (size_t)l * D_ * D_;
    const size_t bo = (size_t)l * D_;
    gemm_qkv<<<gGemm, 128, 0, stream>>>(xbf, wqb + wo, qb + bo, qh, 0, QSCALE);
    gemm_qkv<<<gGemm, 128, 0, stream>>>(xbf, wkb + wo, kb + bo, kh, 0, 1.0f);
    gemm_qkv<<<gGemm, 128, 0, stream>>>(xbf, wvb + wo, vb + bo, vT, 1, 1.0f);

    attn_kernel<<<B_ * H_ * (N_ / 16), 32, 0, stream>>>(
        qh, kh, vT, spatial, attn_bias, sp_emb, sp_emb_rev, abf);

    gemm_res_f32<<<gGemm, 128, 0, stream>>>(abf, wob + wo, ob + bo, x, tmp);
    ln_kernel<<<M_, 256, 0, stream>>>(tmp, ln1w + bo, ln1b + bo, x, xbf);

    gemm_gelu_bf<<<gGemm, 128, 0, stream>>>(xbf, w1b + wo, fc1b + bo, hbf);
    gemm_res_f32<<<gGemm, 128, 0, stream>>>(hbf, w2b + wo, fc2b + bo, x, tmp);
    ln_kernel<<<M_, 256, 0, stream>>>(tmp, ln2w + bo, ln2b + bo,
                                      (l == L_ - 1) ? (float*)d_out : x, xbf);
  }
}
